// MATnet_11605001634055
// MI455X (gfx1250) — matrix operand order repaired and checked by reading, not compiled, not run on hardware
//
#include <hip/hip_runtime.h>
#include <hip/hip_bf16.h>
#include <math.h>

// ---------------------------------------------------------------------------
// MATnet forward for MI455X (gfx1250, wave32, WMMA bf16 w/ fp32 accumulate).
// VOCAB=30000, EMB=300, FEAT=2048, B=256, NQ=32, NW=12, NK=64.
// Outputs: p_emb [8192,300] then k_emb [16384,300], fp32, concatenated.
// ---------------------------------------------------------------------------
#define VOCAB_ 30000
#define EMB_   300
#define PAD_   320                     // bf16 staging row stride (zero-padded)
#define FEAT_  2048
#define BATCH_ 256
#define NQ_    32
#define NW_    12
#define NK_    64
#define MK_    (BATCH_ * NK_)          // 16384
#define MQ_    (BATCH_ * NQ_ * NW_)    // 98304
#define MP_    (BATCH_ * NQ_)          // 8192
#define NTILE_EMB_ 19

typedef __attribute__((ext_vector_type(16))) __bf16 v16bf;
typedef __attribute__((ext_vector_type(8)))  __bf16 v8bf;
typedef __attribute__((ext_vector_type(8)))  float  v8f;

__device__ __forceinline__ __bf16 f2bf(float x) { return (__bf16)x; }

// ---- WMMA fragment loaders (ISA 7.12.2 layouts, wave32) --------------------
// A 16x32 bf16 from fp32 rows (on-the-fly convert; K multiple of 32).
__device__ __forceinline__ v16bf load_a_f32(const float* __restrict__ row,
                                            int kbase, int half) {
  v16bf a;
  const int k0 = kbase + half * 8;
#pragma unroll
  for (int e = 0; e < 8; ++e) {
    a[e]     = f2bf(row[k0 + e]);
    a[e + 8] = f2bf(row[k0 + 16 + e]);
  }
  return a;
}
// A 16x32 bf16 from staged bf16 rows: two aligned 16B chunks.
__device__ __forceinline__ v16bf load_a_bf(const __bf16* __restrict__ row,
                                           int kbase, int half) {
  const int k0 = kbase + half * 8;
  v8bf lo = *(const v8bf*)(row + k0);
  v8bf hi = *(const v8bf*)(row + k0 + 16);
  v16bf a;
#pragma unroll
  for (int e = 0; e < 8; ++e) { a[e] = lo[e]; a[e + 8] = hi[e]; }
  return a;
}
// B 32x16 bf16 from staged bf16 rows: two aligned 16B chunks (lane = column n).
__device__ __forceinline__ v16bf load_b_bf(const __bf16* __restrict__ rowN,
                                           int kbase, int half) {
  const int k0 = kbase + half * 8;
  v8bf lo = *(const v8bf*)(rowN + k0);
  v8bf hi = *(const v8bf*)(rowN + k0 + 16);
  v16bf b;
#pragma unroll
  for (int e = 0; e < 8; ++e) { b[e] = lo[e]; b[e + 8] = hi[e]; }
  return b;
}

#define WMMA_BF16(A, B, C) \
  __builtin_amdgcn_wmma_f32_16x16x32_bf16(false, (A), false, (B), (short)0, (C), false, false)

// ---------------------------------------------------------------------------
// Staging kernels: one-time fp32 -> bf16 conversions with zero padding.
// ---------------------------------------------------------------------------
__global__ void mat_cvt_pad(const float* __restrict__ src, __bf16* __restrict__ dst,
                            int rows, int cols /*=EMB_*/, long total /*rows*PAD_*/) {
  long i = (long)blockIdx.x * blockDim.x + threadIdx.x;
  if (i >= total) return;
  int r = (int)(i / PAD_), c = (int)(i % PAD_);
  dst[i] = (c < cols) ? f2bf(src[(size_t)r * cols + c]) : f2bf(0.0f);
}

__global__ void mat_cvt_wf(const float* __restrict__ Wf, __bf16* __restrict__ wf_bf) {
  long i = (long)blockIdx.x * blockDim.x + threadIdx.x;
  if (i < (long)EMB_ * FEAT_) wf_bf[i] = f2bf(Wf[i]);
}

// Wc = bf16(Wp + eps*Wm) padded to [300,320]; bc = bp + eps*bm (fp32).
__global__ void mat_combine_w(const float* __restrict__ Wp, const float* __restrict__ Wm,
                              const float* __restrict__ bp, const float* __restrict__ bm,
                              __bf16* __restrict__ wc_bf, float* __restrict__ bc) {
  const float eps = 1e-5f;
  long i = (long)blockIdx.x * blockDim.x + threadIdx.x;
  if (i < (long)EMB_ * PAD_) {
    int r = (int)(i / PAD_), c = (int)(i % PAD_);
    wc_bf[i] = (c < EMB_) ? f2bf(Wp[(size_t)r * EMB_ + c] + eps * Wm[(size_t)r * EMB_ + c])
                          : f2bf(0.0f);
  }
  if (i < EMB_) bc[i] = bp[i] + eps * bm[i];
}

// Zero the pad columns [300,320) of the bf16 k_emb copy (GEMM writes n<300).
__global__ void mat_zero_kemb_pad(__bf16* __restrict__ kemb_bf) {
  long i = (long)blockIdx.x * blockDim.x + threadIdx.x;      // [0, MK_*20)
  if (i >= (long)MK_ * (PAD_ - EMB_)) return;
  int r = (int)(i / (PAD_ - EMB_)), c = EMB_ + (int)(i % (PAD_ - EMB_));
  kemb_bf[(size_t)r * PAD_ + c] = f2bf(0.0f);
}

// ---------------------------------------------------------------------------
// k_emb GEMM: feature(16384x2048 fp32, streamed) @ Wf^T(bf16) -> 16x64/wave.
// A fragment converted once per k-step, reused by 4 WMMAs.
// grid = (128, 5), block = 256 (8 waves). Epilogue: +wv[label]+bf, fp32 out
// plus bf16 padded copy for the attention GEMM.
// ---------------------------------------------------------------------------
__global__ void mat_kemb_gemm(const float* __restrict__ feat, const __bf16* __restrict__ wf_bf,
                              const float* __restrict__ bfv,  const float* __restrict__ wv,
                              const int* __restrict__ label,  float* __restrict__ kout,
                              __bf16* __restrict__ kemb_bf) {
  const int lane = threadIdx.x & 31;
  const int wave = threadIdx.x >> 5;
  const int half = lane >> 4;
  const int nl   = lane & 15;
  const int mtile = blockIdx.x * 8 + wave;                   // [0,1024)
  const int nbase = blockIdx.y * 64;                         // 0,64,...,256

  const float* __restrict__ Arow = feat + (size_t)(mtile * 16 + nl) * FEAT_;
  const __bf16* __restrict__ B0;
  const __bf16* __restrict__ B1;
  const __bf16* __restrict__ B2;
  const __bf16* __restrict__ B3;
  {
    int n0 = nbase +  0 + nl; n0 = n0 < EMB_ ? n0 : (EMB_ - 1);
    int n1 = nbase + 16 + nl; n1 = n1 < EMB_ ? n1 : (EMB_ - 1);
    int n2 = nbase + 32 + nl; n2 = n2 < EMB_ ? n2 : (EMB_ - 1);
    int n3 = nbase + 48 + nl; n3 = n3 < EMB_ ? n3 : (EMB_ - 1);
    B0 = wf_bf + (size_t)n0 * FEAT_;  B1 = wf_bf + (size_t)n1 * FEAT_;
    B2 = wf_bf + (size_t)n2 * FEAT_;  B3 = wf_bf + (size_t)n3 * FEAT_;
  }

  v8f c0 = {}, c1 = {}, c2 = {}, c3 = {};
  for (int kb = 0; kb < FEAT_; kb += 32) {
    __builtin_prefetch(Arow + kb + 128, 0, 1);               // global_prefetch_b8
    v16bf a  = load_a_f32(Arow, kb, half);
    v16bf b0 = load_b_bf(B0, kb, half);
    v16bf b1 = load_b_bf(B1, kb, half);
    v16bf b2 = load_b_bf(B2, kb, half);
    v16bf b3 = load_b_bf(B3, kb, half);
    c0 = WMMA_BF16(a, b0, c0);
    c1 = WMMA_BF16(a, b1, c1);
    c2 = WMMA_BF16(a, b2, c2);
    c3 = WMMA_BF16(a, b3, c3);
  }

  int mrow[8], lbl[8];
#pragma unroll
  for (int v = 0; v < 8; ++v) {
    mrow[v] = mtile * 16 + half * 8 + v;
    lbl[v]  = label[mrow[v]];
  }
#pragma unroll
  for (int t = 0; t < 4; ++t) {
    int n = nbase + t * 16 + nl;
    if (n >= EMB_) continue;
    v8f c = (t == 0) ? c0 : (t == 1) ? c1 : (t == 2) ? c2 : c3;
    float bias = bfv[n];
#pragma unroll
    for (int v = 0; v < 8; ++v) {
      float val = c[v] + wv[(size_t)lbl[v] * EMB_ + n] + bias;
      kout[(size_t)mrow[v] * EMB_ + n]    = val;
      kemb_bf[(size_t)mrow[v] * PAD_ + n] = f2bf(val);
    }
  }
}

// ---------------------------------------------------------------------------
// Attention + softmax-over-objects collapsed to qmax = 1/sum(exp(x - max)).
// All-bf16 operands (wv_bf gather rows, kemb_bf), K = 10 unmasked steps.
// grid = (256, 3), block = 256.
// ---------------------------------------------------------------------------
__global__ void mat_attn_qmax(const __bf16* __restrict__ wv_bf, const __bf16* __restrict__ kemb_bf,
                              const int* __restrict__ query, float* __restrict__ qmax) {
  const int lane = threadIdx.x & 31;
  const int wave = threadIdx.x >> 5;
  const int half = lane >> 4;
  const int nl   = lane & 15;
  const int b     = blockIdx.x;
  const int mtile = blockIdx.y * 8 + wave;                   // [0,24)
  const int m     = mtile * 16 + nl;                         // [0,384)

  const int wid = query[b * (NQ_ * NW_) + m];
  const __bf16* __restrict__ Arow = wv_bf + (size_t)wid * PAD_;
  const __bf16* __restrict__ B0 = kemb_bf + ((size_t)b * NK_ +  0 + nl) * PAD_;
  const __bf16* __restrict__ B1 = kemb_bf + ((size_t)b * NK_ + 16 + nl) * PAD_;
  const __bf16* __restrict__ B2 = kemb_bf + ((size_t)b * NK_ + 32 + nl) * PAD_;
  const __bf16* __restrict__ B3 = kemb_bf + ((size_t)b * NK_ + 48 + nl) * PAD_;

  v8f c0 = {}, c1 = {}, c2 = {}, c3 = {};
#pragma unroll
  for (int kb = 0; kb < PAD_; kb += 32) {                    // 10 steps, pad = 0
    v16bf a  = load_a_bf(Arow, kb, half);
    v16bf b0 = load_b_bf(B0, kb, half);
    v16bf b1 = load_b_bf(B1, kb, half);
    v16bf b2 = load_b_bf(B2, kb, half);
    v16bf b3 = load_b_bf(B3, kb, half);
    c0 = WMMA_BF16(a, b0, c0);
    c1 = WMMA_BF16(a, b1, c1);
    c2 = WMMA_BF16(a, b2, c2);
    c3 = WMMA_BF16(a, b3, c3);
  }

  const float scale = 0.05773502691896258f;                  // 1/sqrt(300)
#pragma unroll
  for (int v = 0; v < 8; ++v) {
    float x0 = c0[v] * scale, x1 = c1[v] * scale, x2 = c2[v] * scale, x3 = c3[v] * scale;
    float mx = fmaxf(fmaxf(x0, x1), fmaxf(x2, x3));
#pragma unroll
    for (int off = 1; off <= 8; off <<= 1) mx = fmaxf(mx, __shfl_xor(mx, off, 32));
    float s = __expf(x0 - mx) + __expf(x1 - mx) + __expf(x2 - mx) + __expf(x3 - mx);
#pragma unroll
    for (int off = 1; off <= 8; off <<= 1) s += __shfl_xor(s, off, 32);
    if (nl == 0)
      qmax[(size_t)b * (NQ_ * NW_) + mtile * 16 + half * 8 + v] = 1.0f / s;
  }
}

// ---------------------------------------------------------------------------
// Word softmax: mask query==0 -> -max_float, softmax over 12 words, /5.
// ---------------------------------------------------------------------------
__global__ void mat_word_softmax(const float* __restrict__ qmax, const int* __restrict__ query,
                                 float* __restrict__ wts) {
  const int lane = threadIdx.x & 31;
  const int wave = threadIdx.x >> 5;
  const int row  = blockIdx.x * 8 + wave;                    // [0,8192)
  const float NEG = -3.402823466e38f;

  float v = NEG;
  if (lane < NW_) {
    int q = query[row * NW_ + lane];
    v = (q == 0) ? NEG : qmax[row * NW_ + lane];
  }
  float mx = v;
#pragma unroll
  for (int off = 1; off < 32; off <<= 1) mx = fmaxf(mx, __shfl_xor(mx, off, 32));
  float e = (lane < NW_) ? __expf(v - mx) : 0.0f;
  float s = e;
#pragma unroll
  for (int off = 1; off < 32; off <<= 1) s += __shfl_xor(s, off, 32);
  if (lane < NW_) wts[row * NW_ + lane] = (e / s) * 0.2f;
}

// ---------------------------------------------------------------------------
// p_emb gather (fp32 math from fp32 wv), written as padded bf16 for the
// projection GEMM A operand. One wave per (b,q) row.
// ---------------------------------------------------------------------------
__global__ void mat_pemb_gather(const float* __restrict__ wv, const int* __restrict__ query,
                                const float* __restrict__ wts, __bf16* __restrict__ pemb_bf) {
  const int lane = threadIdx.x & 31;
  const int wave = threadIdx.x >> 5;
  const int row  = blockIdx.x * 8 + wave;                    // [0,8192)

  float w[NW_];
  const float* rows[NW_];
#pragma unroll
  for (int i = 0; i < NW_; ++i) {
    w[i]    = wts[row * NW_ + i];
    rows[i] = wv + (size_t)query[row * NW_ + i] * EMB_;
  }
  for (int d = lane; d < PAD_; d += 32) {
    float acc = 0.0f;
    if (d < EMB_) {
#pragma unroll
      for (int i = 0; i < NW_; ++i) acc = fmaf(w[i], rows[i][d], acc);
    }
    pemb_bf[(size_t)row * PAD_ + d] = f2bf(acc);
  }
}

// ---------------------------------------------------------------------------
// Projection: out = p_emb @ (Wp+eps*Wm)^T + (bp+eps*bm). All-bf16 operands.
// grid = (64, 19), block = 256.
// ---------------------------------------------------------------------------
__global__ void mat_proj_gemm(const __bf16* __restrict__ pemb_bf, const __bf16* __restrict__ wc_bf,
                              const float* __restrict__ bc, float* __restrict__ out) {
  const int lane = threadIdx.x & 31;
  const int wave = threadIdx.x >> 5;
  const int half = lane >> 4;
  const int nl   = lane & 15;
  const int mtile = blockIdx.x * 8 + wave;                   // [0,512)
  const int n     = blockIdx.y * 16 + nl;                    // [0,304)
  const int nEff  = n < EMB_ ? n : (EMB_ - 1);

  const __bf16* __restrict__ Arow = pemb_bf + (size_t)(mtile * 16 + nl) * PAD_;
  const __bf16* __restrict__ Brow = wc_bf   + (size_t)nEff * PAD_;

  v8f c = {};
#pragma unroll
  for (int kb = 0; kb < PAD_; kb += 32) {                    // 10 steps, pad = 0
    v16bf a = load_a_bf(Arow, kb, half);
    v16bf b = load_b_bf(Brow, kb, half);
    c = WMMA_BF16(a, b, c);
  }
  if (n < EMB_) {
    float bias = bc[n];
#pragma unroll
    for (int v = 0; v < 8; ++v) {
      int m = mtile * 16 + half * 8 + v;
      out[(size_t)m * EMB_ + n] = c[v] + bias;
    }
  }
}

// ---------------------------------------------------------------------------
extern "C" void kernel_launch(void* const* d_in, const int* in_sizes, int n_in,
                              void* d_out, int out_size, void* d_ws, size_t ws_size,
                              hipStream_t stream) {
  (void)in_sizes; (void)n_in; (void)out_size; (void)ws_size;

  const float* wv      = (const float*)d_in[0];
  const float* Wf      = (const float*)d_in[1];
  const float* bfv     = (const float*)d_in[2];
  const float* Wp      = (const float*)d_in[3];
  const float* bp      = (const float*)d_in[4];
  const float* Wm      = (const float*)d_in[5];
  const float* bm      = (const float*)d_in[6];
  const float* feature = (const float*)d_in[7];
  const int*   query   = (const int*)  d_in[8];
  const int*   label   = (const int*)  d_in[9];

  float* out_p = (float*)d_out;                              // [8192, 300]
  float* out_k = out_p + (size_t)MP_ * EMB_;                 // [16384, 300]

  // Workspace layout (fp32 first, then 32B-aligned bf16 staging, ~37 MB).
  float*  qmax    = (float*)d_ws;                            // [98304]
  float*  wts     = qmax + MQ_;                              // [98304]
  float*  bc      = wts + MQ_;                               // [320]
  __bf16* wv_bf   = (__bf16*)(bc + PAD_);                    // [30000*320]
  __bf16* kemb_bf = wv_bf   + (size_t)VOCAB_ * PAD_;         // [16384*320]
  __bf16* pemb_bf = kemb_bf + (size_t)MK_ * PAD_;            // [8192*320]
  __bf16* wf_bf   = pemb_bf + (size_t)MP_ * PAD_;            // [300*2048]
  __bf16* wc_bf   = wf_bf   + (size_t)EMB_ * FEAT_;          // [300*320]

  const long nWv = (long)VOCAB_ * PAD_;
  mat_cvt_pad     <<<dim3((unsigned)((nWv + 255) / 256)),              dim3(256), 0, stream>>>(wv, wv_bf, VOCAB_, EMB_, nWv);
  mat_cvt_wf      <<<dim3(((EMB_ * FEAT_) + 255) / 256),               dim3(256), 0, stream>>>(Wf, wf_bf);
  mat_combine_w   <<<dim3(((EMB_ * PAD_) + 255) / 256),                dim3(256), 0, stream>>>(Wp, Wm, bp, bm, wc_bf, bc);
  mat_zero_kemb_pad<<<dim3(((MK_ * (PAD_ - EMB_)) + 255) / 256),       dim3(256), 0, stream>>>(kemb_bf);

  mat_kemb_gemm   <<<dim3(MK_ / 16 / 8, 5),   dim3(256), 0, stream>>>(feature, wf_bf, bfv, wv, label, out_k, kemb_bf);
  mat_attn_qmax   <<<dim3(BATCH_, 3),         dim3(256), 0, stream>>>(wv_bf, kemb_bf, query, qmax);
  mat_word_softmax<<<dim3(MP_ / 8),           dim3(256), 0, stream>>>(qmax, query, wts);
  mat_pemb_gather <<<dim3(MP_ / 8),           dim3(256), 0, stream>>>(wv, query, wts, pemb_bf);
  mat_proj_gemm   <<<dim3(MP_ / 16 / 8, NTILE_EMB_), dim3(256), 0, stream>>>(pemb_bf, wc_bf, bc, out_p);
}
